// PositionAttnModule_6485400617360
// MI455X (gfx1250) — matrix operand order repaired and checked by reading, not compiled, not run on hardware
//
#include <hip/hip_runtime.h>
#include <stdint.h>

#define B_   128
#define T_   256
#define H_   512
#define S_   64
#define C_   6625
#define CPAD 6656   // 104 * 64
#define NBLK 104    // ceil(6625/64)

typedef __attribute__((ext_vector_type(8)))  float  v8f;
typedef __attribute__((ext_vector_type(8)))  __bf16 v8bf;
typedef __attribute__((ext_vector_type(16))) __bf16 v16bf;

static __device__ inline v16bf join16(v8bf lo, v8bf hi) {
  return __builtin_shufflevector(lo, hi, 0,1,2,3,4,5,6,7,8,9,10,11,12,13,14,15);
}

// A-fragment (16x32 bf16, MxK). Lane l holds row m=l&15; halves map to
// K = hi*16 + i  (contiguous, matches B fragments)
static __device__ inline v16bf ldfragA(const __bf16* rowp, int hi) {
  v8bf lo = *(const v8bf*)(rowp + hi * 16);
  v8bf hh = *(const v8bf*)(rowp + hi * 16 + 8);
  return join16(lo, hh);
}
// B-fragment (32x16 bf16, KxN) from [n][k]-transposed storage:
// lane l holds col n=l&15; halves K = hi*16 + i  (contiguous 32B)
static __device__ inline v16bf ldfragB(const __bf16* rowp, int hi) {
  v8bf lo = *(const v8bf*)(rowp + hi * 16);
  v8bf hh = *(const v8bf*)(rowp + hi * 16 + 8);
  return join16(lo, hh);
}

// Async global->LDS 16B copy (CDNA5, ASYNCcnt-tracked). GV mode: 64-bit vaddr.
static __device__ __forceinline__ void async_ld_b128(uint32_t lds_addr,
                                                     const void* gaddr) {
  asm volatile("global_load_async_to_lds_b128 %0, %1, off"
               :: "v"(lds_addr), "v"(gaddr) : "memory");
}
static __device__ __forceinline__ uint32_t lds_u32(const void* p) {
  return (uint32_t)(uintptr_t)p;   // generic shared ptr: low 32 bits = LDS addr
}

// ---------------------------------------------------------------------------
// Kernel 0: Wt[n][k] = bf16(W[k][n]), N padded to 6656 with zeros.
// Reads coalesced along n (128B/wave); writes 16B chunks per row.
// ---------------------------------------------------------------------------
__global__ __launch_bounds__(256)
void wt_convert_kernel(const float* __restrict__ W, __bf16* __restrict__ Wt) {
  const int n = blockIdx.x * 256 + threadIdx.x;    // 0..6655
  const bool valid = (n < C_);
  for (int k = 0; k < H_; k += 8) {
    v8bf v;
    #pragma unroll
    for (int j = 0; j < 8; ++j)
      v[j] = (__bf16)(valid ? W[(size_t)(k + j) * C_ + n] : 0.0f);
    *(v8bf*)&Wt[(size_t)n * H_ + k] = v;
  }
}

// ---------------------------------------------------------------------------
// Kernel 1: per batch b -> scores (WMMA bf16) -> softmax over T -> context
// Block = 128 threads (4 waves); wave w owns S-strip [16w, 16w+16).
// Context written as bf16 to workspace: ctx[(b*64+s)*512 + h]
// ---------------------------------------------------------------------------
__global__ __launch_bounds__(128)
void attn_ctx_kernel(const float* __restrict__ batch_h,
                     const float* __restrict__ pos_emb,
                     __bf16* __restrict__ ctx) {
  __shared__ __align__(16) __bf16 sHb[T_][40];   // phase1 A staging: [t][k32]
  __shared__ __align__(16) __bf16 aT[S_][264];   // attention weights [s][t]
  __shared__ __align__(16) __bf16 sBt[64][40];   // phase2 B staging: [h][t32]

  const int b    = blockIdx.x;
  const int tid  = threadIdx.x;
  const int lane = tid & 31;
  const int wave = tid >> 5;
  const int ln   = lane & 15;
  const int hi   = lane >> 4;
  const int s0   = wave * 16;
  const float* hb = batch_h + (size_t)b * T_ * H_;

  // ---------------- phase 1: scores[t, s-strip] ----------------
  v8f acc[16] = {};
  for (int kk = 0; kk < 16; ++kk) {
    const int kbase = kk * 32;
    __syncthreads();
    {   // stage Hb[0:256][kbase:+32] -> bf16 LDS (coalesced 128B row segments)
      const int r0 = tid >> 2;
      const int cg = (tid & 3) * 8;
      #pragma unroll
      for (int p = 0; p < 8; ++p) {
        const int t = p * 32 + r0;
        const float* g = hb + (size_t)t * H_ + kbase + cg;
        v8bf v;
        #pragma unroll
        for (int j = 0; j < 8; ++j) v[j] = (__bf16)g[j];
        *(v8bf*)&sHb[t][cg] = v;
      }
    }
    __syncthreads();

    // B fragment straight from pos_emb (row-major S x H == [n][k] layout)
    v16bf bfrag;
    {
      const float* g = pos_emb + (size_t)(s0 + ln) * H_ + kbase + hi * 16;
      #pragma unroll
      for (int i = 0; i < 16; ++i) bfrag[i] = (__bf16)g[i];
    }
    #pragma unroll
    for (int tt = 0; tt < 16; ++tt) {
      v16bf afrag = ldfragA(&sHb[tt * 16 + ln][0], hi);
      acc[tt] = __builtin_amdgcn_wmma_f32_16x16x32_bf16(
          false, afrag, false, bfrag, (short)0, acc[tt], false, false);
    }
  }

  // ---------------- softmax over t (column s = s0+ln; partner lane ^16) -----
  {
    float mx = -3.0e38f;
    #pragma unroll
    for (int tt = 0; tt < 16; ++tt)
      #pragma unroll
      for (int r = 0; r < 8; ++r) mx = fmaxf(mx, acc[tt][r]);
    mx = fmaxf(mx, __shfl_xor(mx, 16, 32));
    float sum = 0.f;
    #pragma unroll
    for (int tt = 0; tt < 16; ++tt)
      #pragma unroll
      for (int r = 0; r < 8; ++r) {
        float e = __expf(acc[tt][r] - mx);
        acc[tt][r] = e;
        sum += e;
      }
    sum += __shfl_xor(sum, 16, 32);
    const float inv = 1.0f / sum;
    const int s = s0 + ln;
    #pragma unroll
    for (int tt = 0; tt < 16; ++tt)
      #pragma unroll
      for (int r = 0; r < 8; ++r)
        aT[s][tt * 16 + hi * 8 + r] = (__bf16)(acc[tt][r] * inv);
  }

  // ---------------- phase 2: context[s-strip, :] = a^T x Hb -----------------
  // A' fragments (M=s, K=t=256): 8 resident fragments per wave (own rows only)
  v16bf afr[8];
  #pragma unroll
  for (int k2 = 0; k2 < 8; ++k2)
    afr[k2] = ldfragA(&aT[s0 + ln][k2 * 32], hi);

  for (int nc = 0; nc < 8; ++nc) {
    const int hbase = nc * 64;
    v8f acc2[4] = {};
    for (int k2 = 0; k2 < 8; ++k2) {
      const int tbase = k2 * 32;
      __syncthreads();
      {   // stage Hb[tbase:+32][hbase:+64] transposed -> sBt[h][t]
        const int tt = tid >> 2;
        const int hg = (tid & 3) * 16;
        const float* g = hb + (size_t)(tbase + tt) * H_ + hbase + hg;
        #pragma unroll
        for (int j = 0; j < 16; ++j) sBt[hg + j][tt] = (__bf16)g[j];
      }
      __syncthreads();
      #pragma unroll
      for (int nt = 0; nt < 4; ++nt) {
        v16bf bfr = ldfragB(&sBt[nt * 16 + ln][0], hi);
        acc2[nt] = __builtin_amdgcn_wmma_f32_16x16x32_bf16(
            false, afr[k2], false, bfr, (short)0, acc2[nt], false, false);
      }
    }
    #pragma unroll
    for (int nt = 0; nt < 4; ++nt) {
      const int h = hbase + nt * 16 + ln;
      #pragma unroll
      for (int r = 0; r < 8; ++r) {
        const int s = s0 + hi * 8 + r;
        ctx[((size_t)(b * S_ + s)) * H_ + h] = (__bf16)acc2[nt][r];
      }
    }
  }
}

// ---------------------------------------------------------------------------
// Kernel 2: out = ctx(8192x512 bf16) x Wt^T(bf16) + bias
// 64x64 block tile, 4 waves x (16x64), K in 16 steps of 32.
// Double-buffered LDS, async global->LDS b128 copies (ASYNCcnt pipelined).
// ---------------------------------------------------------------------------
__global__ __launch_bounds__(128)
void gen_gemm_kernel(const __bf16* __restrict__ ctx,
                     const __bf16* __restrict__ Wt,
                     const float* __restrict__ bias,
                     float* __restrict__ out) {
  __shared__ __align__(16) __bf16 sA[2][64][40];   // [buf][m][k32]
  __shared__ __align__(16) __bf16 sB[2][64][40];   // [buf][n][k32]

  const int Nbase = blockIdx.x * 64;
  const int Mbase = blockIdx.y * 64;
  const int tid  = threadIdx.x;
  const int lane = tid & 31;
  const int wave = tid >> 5;
  const int ln   = lane & 15;
  const int hi   = lane >> 4;

  // Per-thread staging slots: row r (0..63), 16-half chunk cg (0 or 16).
  const int r  = tid >> 1;
  const int cg = (tid & 1) * 16;
  const __bf16* gA = ctx + (size_t)(Mbase + r) * H_ + cg;
  const __bf16* gB = Wt  + (size_t)(Nbase + r) * H_ + cg;
  const uint32_t lA[2] = { lds_u32(&sA[0][r][cg]), lds_u32(&sA[1][r][cg]) };
  const uint32_t lB[2] = { lds_u32(&sB[0][r][cg]), lds_u32(&sB[1][r][cg]) };

  // issue 4 async b128 copies for K-step kk into buffer `buf`
  auto issue = [&](int buf, int kk) {
    const int kbase = kk * 32;
    async_ld_b128(lA[buf],      gA + kbase);
    async_ld_b128(lA[buf] + 16, gA + kbase + 8);
    async_ld_b128(lB[buf],      gB + kbase);
    async_ld_b128(lB[buf] + 16, gB + kbase + 8);
  };

  v8f acc[4] = {};
  issue(0, 0);
  #pragma unroll
  for (int kk = 0; kk < 16; ++kk) {
    const int cur = kk & 1;
    if (kk < 15) {
      issue(cur ^ 1, kk + 1);
      asm volatile("s_wait_asynccnt 0x4" ::: "memory");  // current step landed
    } else {
      asm volatile("s_wait_asynccnt 0x0" ::: "memory");
    }
    __syncthreads();            // all waves' copies visible in LDS
    v16bf af = ldfragA(&sA[cur][wave * 16 + ln][0], hi);
    #pragma unroll
    for (int nt = 0; nt < 4; ++nt) {
      v16bf bf = ldfragB(&sB[cur][nt * 16 + ln][0], hi);
      acc[nt] = __builtin_amdgcn_wmma_f32_16x16x32_bf16(
          false, af, false, bf, (short)0, acc[nt], false, false);
    }
    __syncthreads();            // done reading `cur` before it is overwritten
  }

  // epilogue: bias + fp32 store (rows of 64B per VGPR -> coalesced)
  #pragma unroll
  for (int nt = 0; nt < 4; ++nt) {
    const int n = Nbase + nt * 16 + ln;
    if (n < C_) {
      const float bv = bias[n];
      #pragma unroll
      for (int rr = 0; rr < 8; ++rr) {
        const int m = Mbase + wave * 16 + hi * 8 + rr;
        out[(size_t)m * C_ + n] = acc[nt][rr] + bv;
      }
    }
  }
}

extern "C" void kernel_launch(void* const* d_in, const int* in_sizes, int n_in,
                              void* d_out, int out_size, void* d_ws, size_t ws_size,
                              hipStream_t stream) {
  const float* batch_h = (const float*)d_in[0];
  const float* pos_emb = (const float*)d_in[1];
  const float* W_gen   = (const float*)d_in[2];
  const float* b_gen   = (const float*)d_in[3];
  float*  out = (float*)d_out;

  __bf16* ctx = (__bf16*)d_ws;                       // 8 MB
  __bf16* Wt  = ctx + (size_t)B_ * S_ * H_;          // 6.8 MB (padded, transposed)

  wt_convert_kernel<<<dim3(CPAD / 256), dim3(256), 0, stream>>>(W_gen, Wt);
  attn_ctx_kernel<<<dim3(B_), dim3(128), 0, stream>>>(batch_h, pos_emb, ctx);
  gen_gemm_kernel<<<dim3(NBLK, (B_ * S_) / 64), dim3(128), 0, stream>>>(
      ctx, Wt, b_gen, out);
}
